// GroupedQueryAttention_75505525064719
// MI455X (gfx1250) — matrix operand order repaired and checked by reading, not compiled, not run on hardware
//
#include <hip/hip_runtime.h>

typedef __bf16 bf16;
typedef __attribute__((ext_vector_type(16))) __bf16 v16bf;
typedef __attribute__((ext_vector_type(8)))  float  v8f;

// Problem constants (fixed by the reference)
constexpr int Bc   = 2;
constexpr int Sc   = 2048;
constexpr int Dc   = 2048;
constexpr int Hc   = 16;
constexpr int KVHc = 4;
constexpr int Gc   = 4;      // H / KVH
constexpr int HDc  = 128;
constexpr int Mrows = Bc * Sc;      // 4096 tokens
constexpr int NQ   = Hc * HDc;      // 2048
constexpr int NKV  = KVHc * HDc;    // 512

__device__ __forceinline__ bf16 f2bf(float f) {
  unsigned u = __float_as_uint(f);
  u += 0x7FFFu + ((u >> 16) & 1u);           // round-to-nearest-even
  unsigned short h = (unsigned short)(u >> 16);
  bf16 r;
  __builtin_memcpy(&r, &h, 2);
  return r;
}
__device__ __forceinline__ float bf2f(bf16 b) {
  unsigned short h;
  __builtin_memcpy(&h, &b, 2);
  unsigned u = ((unsigned)h) << 16;
  return __uint_as_float(u);
}

// ---------------------------------------------------------------- casts
__global__ void cast_f32_bf16(const float* __restrict__ src, bf16* __restrict__ dst, int n) {
  int i = blockIdx.x * blockDim.x + threadIdx.x;
  if (i < n) dst[i] = f2bf(src[i]);
}

// src [R, C] f32 row-major -> dst [C, R] bf16 row-major (weight transpose)
__global__ void cast_transpose_f32_bf16(const float* __restrict__ src, bf16* __restrict__ dst,
                                        int R, int C) {
  int i = blockIdx.x * blockDim.x + threadIdx.x;
  if (i >= R * C) return;
  int c = i % C, r = i / C;
  dst[(size_t)c * R + r] = f2bf(src[(size_t)r * C + c]);
}

// ---------------------------------------------------------------- tile loaders
// A 16x32 tile, ISA layout. base must already point at row(l)*K + hw*8.
__device__ __forceinline__ v16bf load_a_tile(const bf16* __restrict__ base, int k0) {
  v16bf a;
  const bf16* p = base + k0;
#pragma unroll
  for (int e = 0; e < 8; ++e) a[e] = p[e];
#pragma unroll
  for (int e = 0; e < 8; ++e) a[8 + e] = p[16 + e];
  return a;
}
// B 32x16 tile from a [N,K] (transposed) matrix: two 8-element runs at p and p+16.
__device__ __forceinline__ v16bf load_b_contig(const bf16* __restrict__ p) {
  v16bf b;
#pragma unroll
  for (int e = 0; e < 16; ++e) b[e] = p[(e & 7) + ((e >> 3) << 4)];
  return b;
}

// ---------------------------------------------------------------- bf16 WMMA GEMM (TN)
// C[M,N] = A[M,K] * BT[N,K]^T. Each wave: 32x64 output tile, double-buffered K loop.
template <typename OutT>
__device__ __forceinline__ void store_elem(OutT* C, size_t idx, float v);
template <> __device__ __forceinline__ void store_elem<float>(float* C, size_t idx, float v) { C[idx] = v; }
template <> __device__ __forceinline__ void store_elem<bf16>(bf16* C, size_t idx, float v) { C[idx] = f2bf(v); }

template <typename OutT>
__global__ __launch_bounds__(128)
void gemm_bf16_wmma_tn(const bf16* __restrict__ A, const bf16* __restrict__ BT,
                       OutT* __restrict__ C, int M, int N, int K) {
  const int wavesPerBlock = blockDim.x >> 5;
  const int wave = blockIdx.x * wavesPerBlock + (threadIdx.x >> 5);
  const int lane = threadIdx.x & 31;
  const int hw   = lane >> 4;
  const int l    = lane & 15;

  const int ntiles = N >> 6;               // 64-wide column tiles
  const int tiles  = (M >> 5) * ntiles;    // 32-tall row tiles
  if (wave >= tiles) return;
  const int m0 = (wave / ntiles) << 5;
  const int n0 = (wave % ntiles) << 6;

  const bf16* arow0 = A + (size_t)(m0 + l) * K + hw * 8;
  const bf16* arow1 = A + (size_t)(m0 + 16 + l) * K + hw * 8;
  const bf16* brow[4];
#pragma unroll
  for (int t = 0; t < 4; ++t) brow[t] = BT + (size_t)(n0 + t * 16 + l) * K + hw * 8;

  const v8f vzero = {0.f, 0.f, 0.f, 0.f, 0.f, 0.f, 0.f, 0.f};
  v8f acc[2][4];
#pragma unroll
  for (int i = 0; i < 2; ++i)
#pragma unroll
    for (int t = 0; t < 4; ++t) acc[i][t] = vzero;

  // prime
  v16bf a0c = load_a_tile(arow0, 0);
  v16bf a1c = load_a_tile(arow1, 0);
  v16bf bc[4];
#pragma unroll
  for (int t = 0; t < 4; ++t) bc[t] = load_b_contig(brow[t]);

  for (int k0 = 32; k0 < K; k0 += 32) {
    // deep-prefetch the HBM-resident A stream (gfx1250 global_prefetch_b8)
    if ((k0 & 255) == 32) {
      __builtin_prefetch(arow0 + k0 + 256, 0, 0);
      __builtin_prefetch(arow1 + k0 + 256, 0, 0);
    }
    // issue next-iteration loads before consuming current tiles
    v16bf a0n = load_a_tile(arow0, k0);
    v16bf a1n = load_a_tile(arow1, k0);
    v16bf bn[4];
#pragma unroll
    for (int t = 0; t < 4; ++t) bn[t] = load_b_contig(brow[t] + k0);

#pragma unroll
    for (int t = 0; t < 4; ++t) {
      acc[0][t] = __builtin_amdgcn_wmma_f32_16x16x32_bf16(false, a0c, false, bc[t],
                                                          (short)0, acc[0][t], false, false);
      acc[1][t] = __builtin_amdgcn_wmma_f32_16x16x32_bf16(false, a1c, false, bc[t],
                                                          (short)0, acc[1][t], false, false);
    }
    a0c = a0n; a1c = a1n;
#pragma unroll
    for (int t = 0; t < 4; ++t) bc[t] = bn[t];
  }
#pragma unroll
  for (int t = 0; t < 4; ++t) {
    acc[0][t] = __builtin_amdgcn_wmma_f32_16x16x32_bf16(false, a0c, false, bc[t],
                                                        (short)0, acc[0][t], false, false);
    acc[1][t] = __builtin_amdgcn_wmma_f32_16x16x32_bf16(false, a1c, false, bc[t],
                                                        (short)0, acc[1][t], false, false);
  }

#pragma unroll
  for (int i = 0; i < 2; ++i)
#pragma unroll
    for (int t = 0; t < 4; ++t)
#pragma unroll
      for (int r = 0; r < 8; ++r) {
        const size_t m = (size_t)(m0 + i * 16 + r + 8 * hw);
        const size_t n = (size_t)(n0 + t * 16 + l);
        store_elem<OutT>(C, m * (size_t)N + n, acc[i][t][r]);
      }
}

// ---------------------------------------------------------------- RoPE + relayout
__global__ void rope_q_kernel(const bf16* __restrict__ Qp, bf16* __restrict__ Qh) {
  int i = blockIdx.x * blockDim.x + threadIdx.x;
  if (i >= Bc * Sc * Hc * (HDc / 2)) return;
  int d = i & 63; int t = i >> 6;
  int h = t % Hc; t /= Hc;
  int s = t % Sc; int b = t / Sc;
  float inv = __expf(-(float)d * (9.210340371976184f / 64.0f)); // 10000^(-2d/128)
  float ang = (float)s * inv;
  float sn, cs;
  sincosf(ang, &sn, &cs);
  size_t src = (size_t)(b * Sc + s) * NQ + h * HDc;
  float x1 = bf2f(Qp[src + d]);
  float x2 = bf2f(Qp[src + 64 + d]);
  size_t dst = ((size_t)(b * Hc + h) * Sc + s) * HDc;
  Qh[dst + d]      = f2bf(x1 * cs - x2 * sn);
  Qh[dst + 64 + d] = f2bf(x1 * sn + x2 * cs);
}

__global__ void rope_k_kernel(const bf16* __restrict__ Kp, bf16* __restrict__ Kh) {
  int i = blockIdx.x * blockDim.x + threadIdx.x;
  if (i >= Bc * Sc * KVHc * (HDc / 2)) return;
  int d = i & 63; int t = i >> 6;
  int kh = t % KVHc; t /= KVHc;
  int s = t % Sc; int b = t / Sc;
  float inv = __expf(-(float)d * (9.210340371976184f / 64.0f));
  float ang = (float)s * inv;
  float sn, cs;
  sincosf(ang, &sn, &cs);
  size_t src = (size_t)(b * Sc + s) * NKV + kh * HDc;
  float x1 = bf2f(Kp[src + d]);
  float x2 = bf2f(Kp[src + 64 + d]);
  size_t dst = ((size_t)(b * KVHc + kh) * Sc + s) * HDc;
  Kh[dst + d]      = f2bf(x1 * cs - x2 * sn);
  Kh[dst + 64 + d] = f2bf(x1 * sn + x2 * cs);
}

// Vp [B*S, KVH*HD] -> Vt [B, KVH, HD, S]  (keys contiguous for P@V B-tiles)
__global__ void v_relayout_kernel(const bf16* __restrict__ Vp, bf16* __restrict__ Vt) {
  int i = blockIdx.x * blockDim.x + threadIdx.x;
  if (i >= Bc * Sc * KVHc * HDc) return;
  int d = i % HDc; int t = i / HDc;
  int kh = t % KVHc; t /= KVHc;
  int s = t % Sc; int b = t / Sc;
  Vt[((size_t)(b * KVHc + kh) * HDc + d) * Sc + s] =
      Vp[(size_t)(b * Sc + s) * NKV + kh * HDc + d];
}

// ---------------------------------------------------------------- flash attention
template <bool MASK>
__device__ __forceinline__ void attn_block(
    int s0, int q0, int hw, int l,
    const bf16* __restrict__ Kbase,    // Kh + (b*KVH+kvh)*S*HD
    const bf16* __restrict__ Vtbase,   // Vt + (b*KVH+kvh)*HD*S
    const v16bf* __restrict__ qa, v8f* __restrict__ o,
    float* __restrict__ mrow, float* __restrict__ lrow, bf16* __restrict__ pl) {
  const float scale = 0.08838834764831845f;   // 128^-0.5
  const v8f vzero = {0.f, 0.f, 0.f, 0.f, 0.f, 0.f, 0.f, 0.f};

  // Load all K B-tiles for this 32-key block (two 16B runs per lane)
  v16bf kt[2][4];
#pragma unroll
  for (int tt = 0; tt < 2; ++tt) {
    const bf16* kp = Kbase + (size_t)(s0 + tt * 16 + l) * HDc + hw * 8;
#pragma unroll
    for (int c = 0; c < 4; ++c)
#pragma unroll
      for (int e = 0; e < 16; ++e) kt[tt][c][e] = kp[c * 32 + (e & 7) + ((e >> 3) << 4)];
  }

  v8f st[2] = {vzero, vzero};
#pragma unroll
  for (int tt = 0; tt < 2; ++tt)
#pragma unroll
    for (int c = 0; c < 4; ++c)
      st[tt] = __builtin_amdgcn_wmma_f32_16x16x32_bf16(false, qa[c], false, kt[tt][c],
                                                       (short)0, st[tt], false, false);

  // Prefetch V B-tiles now; softmax VALU below hides their latency.
  v16bf vt[8];
#pragma unroll
  for (int j = 0; j < 8; ++j) {
    const bf16* vp = Vtbase + (size_t)(j * 16 + l) * Sc + s0 + hw * 8;
#pragma unroll
    for (int e = 0; e < 16; ++e) vt[j][e] = vp[(e & 7) + ((e >> 3) << 4)];
  }

  float sc2[2][8];
#pragma unroll
  for (int tt = 0; tt < 2; ++tt) {
    const int kpos = s0 + tt * 16 + l;
#pragma unroll
    for (int r = 0; r < 8; ++r) {
      float s = st[tt][r] * scale;
      if (MASK) {
        const int qpos = q0 + r + 8 * hw;
        s = (kpos > qpos) ? -3.0e38f : s;
      }
      sc2[tt][r] = s;
    }
  }

  // Online softmax: queries in vector elems, keys across 16-lane group
#pragma unroll
  for (int r = 0; r < 8; ++r) {
    float rm = fmaxf(sc2[0][r], sc2[1][r]);
#pragma unroll
    for (int off = 1; off < 16; off <<= 1) rm = fmaxf(rm, __shfl_xor(rm, off, 32));
    const float mn = fmaxf(mrow[r], rm);
    const float alpha = __expf(mrow[r] - mn);
    const float p0 = __expf(sc2[0][r] - mn);
    const float p1 = __expf(sc2[1][r] - mn);
    pl[(r + 8 * hw) * 32 + l]      = f2bf(p0);
    pl[(r + 8 * hw) * 32 + 16 + l] = f2bf(p1);
    float ps = p0 + p1;
#pragma unroll
    for (int off = 1; off < 16; off <<= 1) ps += __shfl_xor(ps, off, 32);
    lrow[r] = lrow[r] * alpha + ps;
    mrow[r] = mn;
#pragma unroll
    for (int j = 0; j < 8; ++j) o[j][r] *= alpha;
  }

  // Reread P in A-matrix layout (per-wave LDS slice; DS ops in-order per wave)
  v16bf pa;
#pragma unroll
  for (int e = 0; e < 8; ++e) pa[e] = pl[l * 32 + hw * 8 + e];
#pragma unroll
  for (int e = 0; e < 8; ++e) pa[8 + e] = pl[l * 32 + 16 + hw * 8 + e];

#pragma unroll
  for (int j = 0; j < 8; ++j)
    o[j] = __builtin_amdgcn_wmma_f32_16x16x32_bf16(false, pa, false, vt[j],
                                                   (short)0, o[j], false, false);
}

__global__ __launch_bounds__(128)
void gqa_attention(const bf16* __restrict__ Qh, const bf16* __restrict__ Kh,
                   const bf16* __restrict__ Vt, bf16* __restrict__ AO) {
  __shared__ bf16 plds_all[4][16 * 32];
  const int waveInBlock = threadIdx.x >> 5;
  const int wave = blockIdx.x * 4 + waveInBlock;
  const int lane = threadIdx.x & 31;
  const int hw = lane >> 4;
  const int l = lane & 15;

  const int QT = Sc / 16;
  int t = wave;
  const int qt = t % QT; t /= QT;
  const int h  = t % Hc; t /= Hc;
  const int b  = t;
  if (b >= Bc) return;
  const int kvh = h / Gc;
  const int q0  = qt * 16;

  const bf16* Kbase  = Kh + ((size_t)(b * KVHc + kvh) * Sc) * HDc;
  const bf16* Vtbase = Vt + ((size_t)(b * KVHc + kvh) * HDc) * Sc;

  // Preload 16x128 Q tile as four A-matrices
  v16bf qa[4];
  {
    const size_t qbase = ((size_t)(b * Hc + h) * Sc + q0) * HDc;
#pragma unroll
    for (int c = 0; c < 4; ++c) {
      const bf16* arow = Qh + qbase + (size_t)l * HDc + c * 32 + hw * 8;
#pragma unroll
      for (int e = 0; e < 8; ++e) qa[c][e] = arow[e];
#pragma unroll
      for (int e = 0; e < 8; ++e) qa[c][8 + e] = arow[16 + e];
    }
  }

  const v8f vzero = {0.f, 0.f, 0.f, 0.f, 0.f, 0.f, 0.f, 0.f};
  v8f o[8];
#pragma unroll
  for (int j = 0; j < 8; ++j) o[j] = vzero;
  float mrow[8], lrow[8];
#pragma unroll
  for (int r = 0; r < 8; ++r) { mrow[r] = -3.0e38f; lrow[r] = 0.f; }

  bf16* pl = plds_all[waveInBlock];

  const int nfull = q0 >> 5;   // blocks fully below the diagonal
  for (int kb = 0; kb < nfull; ++kb)
    attn_block<false>(kb * 32, q0, hw, l, Kbase, Vtbase, qa, o, mrow, lrow, pl);
  attn_block<true>(nfull * 32, q0, hw, l, Kbase, Vtbase, qa, o, mrow, lrow, pl);

  // Normalize (8 reciprocals instead of 64 divides) and store token-major
  float invl[8];
#pragma unroll
  for (int r = 0; r < 8; ++r) invl[r] = 1.0f / lrow[r];
#pragma unroll
  for (int j = 0; j < 8; ++j)
#pragma unroll
    for (int r = 0; r < 8; ++r) {
      const float val = o[j][r] * invl[r];
      const size_t row = (size_t)b * Sc + q0 + r + 8 * hw;
      AO[row * NQ + (size_t)h * HDc + j * 16 + l] = f2bf(val);
    }
}

// ---------------------------------------------------------------- host launcher
extern "C" void kernel_launch(void* const* d_in, const int* in_sizes, int n_in,
                              void* d_out, int out_size, void* d_ws, size_t ws_size,
                              hipStream_t stream) {
  const float* x  = (const float*)d_in[0];
  const float* Wq = (const float*)d_in[1];
  const float* Wk = (const float*)d_in[2];
  const float* Wv = (const float*)d_in[3];
  const float* Wo = (const float*)d_in[4];
  float* out = (float*)d_out;

  char* w = (char*)d_ws;
  size_t off = 0;
  auto take = [&](size_t nbytes) -> char* {
    char* p = w + off;
    off += (nbytes + 255) & ~(size_t)255;
    return p;
  };
  bf16* xb   = (bf16*)take((size_t)Mrows * Dc * 2);
  bf16* WqT  = (bf16*)take((size_t)Dc * NQ * 2);    // [NQ, D]
  bf16* WkT  = (bf16*)take((size_t)Dc * NKV * 2);   // [NKV, D]
  bf16* WvT  = (bf16*)take((size_t)Dc * NKV * 2);   // [NKV, D]
  bf16* WoT  = (bf16*)take((size_t)NQ * Dc * 2);    // [D, NQ]
  bf16* Qp   = (bf16*)take((size_t)Mrows * NQ * 2);
  bf16* Kp   = (bf16*)take((size_t)Mrows * NKV * 2);
  bf16* Vp   = (bf16*)take((size_t)Mrows * NKV * 2);
  bf16* Qh   = (bf16*)take((size_t)Mrows * NQ * 2);
  bf16* Kh   = (bf16*)take((size_t)Mrows * NKV * 2);
  bf16* Vt   = (bf16*)take((size_t)Mrows * NKV * 2);
  bf16* AO   = (bf16*)take((size_t)Mrows * NQ * 2);

  auto cgrid = [](int n) { return (n + 255) / 256; };
  cast_f32_bf16<<<cgrid(Mrows * Dc), 256, 0, stream>>>(x, xb, Mrows * Dc);
  cast_transpose_f32_bf16<<<cgrid(Dc * NQ), 256, 0, stream>>>(Wq, WqT, Dc, NQ);
  cast_transpose_f32_bf16<<<cgrid(Dc * NKV), 256, 0, stream>>>(Wk, WkT, Dc, NKV);
  cast_transpose_f32_bf16<<<cgrid(Dc * NKV), 256, 0, stream>>>(Wv, WvT, Dc, NKV);
  cast_transpose_f32_bf16<<<cgrid(NQ * Dc), 256, 0, stream>>>(Wo, WoT, NQ, Dc);

  auto ggrid = [](int M, int N) { return ((M / 32) * (N / 64) + 3) / 4; };
  gemm_bf16_wmma_tn<bf16><<<ggrid(Mrows, NQ), 128, 0, stream>>>(xb, WqT, Qp, Mrows, NQ, Dc);
  gemm_bf16_wmma_tn<bf16><<<ggrid(Mrows, NKV), 128, 0, stream>>>(xb, WkT, Kp, Mrows, NKV, Dc);
  gemm_bf16_wmma_tn<bf16><<<ggrid(Mrows, NKV), 128, 0, stream>>>(xb, WvT, Vp, Mrows, NKV, Dc);

  rope_q_kernel<<<cgrid(Bc * Sc * Hc * 64), 256, 0, stream>>>(Qp, Qh);
  rope_k_kernel<<<cgrid(Bc * Sc * KVHc * 64), 256, 0, stream>>>(Kp, Kh);
  v_relayout_kernel<<<cgrid(Bc * Sc * KVHc * HDc), 256, 0, stream>>>(Vp, Vt);

  const int attnWaves = Bc * Hc * (Sc / 16);        // 4096 waves
  gqa_attention<<<attnWaves / 4, 128, 0, stream>>>(Qh, Kh, Vt, AO);

  gemm_bf16_wmma_tn<float><<<ggrid(Mrows, Dc), 128, 0, stream>>>(AO, WoT, out, Mrows, Dc, NQ);
}
